// GCGRUODECell_32933809225773
// MI455X (gfx1250) — matrix operand order repaired and checked by reading, not compiled, not run on hardware
//
#include <hip/hip_runtime.h>
#include <math.h>

// ---------------------------------------------------------------------------
// GC-GRU ODE cell for MI455X (gfx1250): one workgroup (8 wave32) per batch
// item; five 128x128x128 matmuls via v_wmma_f32_16x16x32_bf16 out of LDS.
// ---------------------------------------------------------------------------

typedef __bf16 bf16;
typedef __attribute__((ext_vector_type(16))) __bf16 v16bf;
typedef __attribute__((ext_vector_type(8)))  __bf16 v8bf;
typedef __attribute__((ext_vector_type(4)))  __bf16 v4bf;
typedef __attribute__((ext_vector_type(8)))  float  v8f;

#define NDIM 128   // nodes per graph
#define DDIM 128   // feature dim

// XOR-swizzled LDS index for a logical (row, col) element of a 128x128 bf16
// matrix. Swizzle permutes 8-element blocks across banks so that strided
// row accesses (WMMA fragment loads: 16 lanes reading 16 different rows at
// the same column block) hit 16 distinct bank groups instead of 1.
// All vector accesses in this kernel are 8-element-block aligned, so the
// XOR (which only touches column bits [6:3]) preserves their alignment.
__device__ __forceinline__ int swi(int r, int c) {
  return r * 128 + (c ^ ((r & 15) << 3));
}

__device__ __forceinline__ v16bf cat8(v8bf lo, v8bf hi) {
  return __builtin_shufflevector(lo, hi, 0, 1, 2, 3, 4, 5, 6, 7,
                                         8, 9, 10, 11, 12, 13, 14, 15);
}

// Load A-operand fragment (16x32 bf16) for tile row m0, k-block k0 from a
// row-major swizzled LDS matrix P.
//   lanes 0-15 : M = lane,     K = k0+0..7   (v0-3),  k0+16..23 (v4-7)
//   lanes16-31 : M = lane-16,  K = k0+8..15  (v0-3),  k0+24..31 (v4-7)
__device__ __forceinline__ v16bf load_afrag(const bf16* P, int lane, int m0, int k0) {
  const int r  = lane & 15;
  const int hl = lane >> 4;
  v8bf lo = *(const v8bf*)&P[swi(m0 + r, k0 + hl * 8)];
  v8bf hi = *(const v8bf*)&P[swi(m0 + r, k0 + 16 + hl * 8)];
  return cat8(lo, hi);
}

// Load B-operand fragment (32x16 bf16) for tile col n0, k-block k0 from a
// column-major (i.e. Q^T row-major) swizzled LDS matrix QT.
//   lanes 0-15 : N = lane,     K = k0+0..7   (v0-3),  k0+16..23 (v4-7)
//   lanes16-31 : N = lane-16,  K = k0+8..15  (v0-3),  k0+24..31 (v4-7)
__device__ __forceinline__ v16bf load_bfrag(const bf16* QT, int lane, int n0, int k0) {
  const int r  = lane & 15;
  const int hl = lane >> 4;
  v8bf lo = *(const v8bf*)&QT[swi(n0 + r, k0 + hl * 8)];
  v8bf hi = *(const v8bf*)&QT[swi(n0 + r, k0 + 16 + hl * 8)];
  return cat8(lo, hi);
}

// One 16-row output strip of a 128x128 = (128x128)@(128x128) matmul.
// P: A-operand row-major swizzled; QT: B-operand column-major swizzled.
// epi(nt, n0, acc) consumes each 16x16 f32 tile.
// C/D layout: acc[v] <-> (M = m0 + v + 8*(lane>>4), N = n0 + (lane&15)).
template <class EPI>
__device__ __forceinline__ void mm_strip(const bf16* P, const bf16* QT,
                                         int lane, int m0, EPI epi) {
  v16bf afr[4];
#pragma unroll
  for (int kk = 0; kk < 4; ++kk)
    afr[kk] = load_afrag(P, lane, m0, kk * 32);

#pragma unroll
  for (int nt = 0; nt < 8; ++nt) {
    const int n0 = nt * 16;
    v8f acc = {0.f, 0.f, 0.f, 0.f, 0.f, 0.f, 0.f, 0.f};
#pragma unroll
    for (int kk = 0; kk < 4; ++kk) {
      v16bf bfr = load_bfrag(QT, lane, n0, kk * 32);
      acc = __builtin_amdgcn_wmma_f32_16x16x32_bf16(
          /*neg_a=*/false, afr[kk], /*neg_b=*/false, bfr,
          /*c_mod=*/(short)0, acc, /*reuse_a=*/false, /*reuse_b=*/false);
    }
    epi(nt, n0, acc);
  }
}

// Cooperative stage of a 128x128 fp32 global matrix into a transposed
// (column-major) swizzled bf16 LDS matrix: dst[c][r] = src[r][c].
__device__ __forceinline__ void stage_transposed(bf16* dst, const float* src, int tid) {
#pragma unroll 4
  for (int i = tid; i < 4096; i += 256) {
    float4 f = ((const float4*)src)[i];      // src elements 4i .. 4i+3
    int r = (4 * i) >> 7;
    int c = (4 * i) & 127;
    dst[swi(c + 0, r)] = (bf16)f.x;
    dst[swi(c + 1, r)] = (bf16)f.y;
    dst[swi(c + 2, r)] = (bf16)f.z;
    dst[swi(c + 3, r)] = (bf16)f.w;
  }
}

__device__ __forceinline__ float sigmoidf_(float x) {
  return 1.0f / (1.0f + __expf(-x));
}

__global__ __launch_bounds__(256, 1)
void gcgru_cell_kernel(const float* __restrict__ h, const float* __restrict__ A,
                       const float* __restrict__ Wr, const float* __restrict__ br,
                       const float* __restrict__ Wz, const float* __restrict__ bz,
                       const float* __restrict__ Wh, const float* __restrict__ bh,
                       float* __restrict__ out) {
  // 160 KB of LDS: one workgroup per WGP half; all matmul operands resident.
  __shared__ alignas(16) bf16 sA[128 * 128];    // A, row-major (A-operand)
  __shared__ alignas(16) bf16 sHT[128 * 128];   // h^T, [feat][node] (B-operand)
  __shared__ alignas(16) bf16 sWT[128 * 128];   // rotating W^T (B-operand)
  __shared__ alignas(16) bf16 sAgg[128 * 128];  // Agg / Agg2, row-major (A-operand)
  __shared__ alignas(16) bf16 sRHT[128 * 128];  // (r*h)^T (B-operand)

  const int tid  = threadIdx.x;
  const int lane = tid & 31;
  const int wave = tid >> 5;
  const int m0   = wave * 16;          // this wave's 16-row output strip
  const int cn   = lane & 15;          // N within tile
  const int mb   = m0 + 8 * (lane >> 4);  // M base for this lane's 8 values

  const size_t bofs = (size_t)blockIdx.x * (NDIM * DDIM);
  const float* hB   = h + bofs;
  const float* AB   = A + bofs;
  float*       outB = out + bofs;

  // -------- stage A (row-major bf16) --------
#pragma unroll 4
  for (int i = tid; i < 4096; i += 256) {
    float4 f = ((const float4*)AB)[i];
    int r = (4 * i) >> 7;
    int c = (4 * i) & 127;
    v4bf p = {(bf16)f.x, (bf16)f.y, (bf16)f.z, (bf16)f.w};
    *(v4bf*)&sA[swi(r, c)] = p;        // 4-elem block stays inside an 8-block
  }
  stage_transposed(sHT, hB, tid);      // h^T
  stage_transposed(sWT, Wr, tid);      // W_r^T
  __syncthreads();

  // -------- MM1: Agg = A @ h -> sAgg (row-major bf16) --------
  mm_strip(sA, sHT, lane, m0, [&](int, int n0, v8f c) {
    const int n = n0 + cn;
#pragma unroll
    for (int v = 0; v < 8; ++v) sAgg[swi(mb + v, n)] = (bf16)c[v];
  });
  __syncthreads();

  // -------- MM2: R = Agg @ W_r ; rh = sigmoid(R)*h -> sRHT --------
  mm_strip(sAgg, sWT, lane, m0, [&](int, int n0, v8f c) {
    const int n = n0 + cn;
    const float bias = br[n];
    v8bf h8 = *(const v8bf*)&sHT[swi(n, mb)];   // h[mb..mb+7, n]
    v8bf rh;
#pragma unroll
    for (int v = 0; v < 8; ++v) {
      float r = sigmoidf_(c[v] + bias);
      rh[v] = (bf16)(r * (float)h8[v]);
    }
    *(v8bf*)&sRHT[swi(n, mb)] = rh;
  });
  __syncthreads();

  stage_transposed(sWT, Wz, tid);      // rotate weights: W_z^T
  __syncthreads();

  // -------- MM3: Z = Agg @ W_z ; z kept in registers --------
  v8f zsig[8];
  mm_strip(sAgg, sWT, lane, m0, [&](int nt, int n0, v8f c) {
    const float bias = bz[n0 + cn];
    v8f zz;
#pragma unroll
    for (int v = 0; v < 8; ++v) zz[v] = sigmoidf_(c[v] + bias);
    zsig[nt] = zz;
  });
  __syncthreads();

  stage_transposed(sWT, Wh, tid);      // rotate weights: W_h^T

  // -------- MM4: Agg2 = A @ rh -> sAgg (reuse) --------
  mm_strip(sA, sRHT, lane, m0, [&](int, int n0, v8f c) {
    const int n = n0 + cn;
#pragma unroll
    for (int v = 0; v < 8; ++v) sAgg[swi(mb + v, n)] = (bf16)c[v];
  });
  __syncthreads();

  // -------- MM5: U = Agg2 @ W_h ; dh = (1-z)*(tanh(U)-h) -> out --------
  mm_strip(sAgg, sWT, lane, m0, [&](int nt, int n0, v8f c) {
    const int n = n0 + cn;
    const float bias = bh[n];
    v8bf h8 = *(const v8bf*)&sHT[swi(n, mb)];
    v8f zz = zsig[nt];
#pragma unroll
    for (int v = 0; v < 8; ++v) {
      float u = tanhf(c[v] + bias);
      outB[(mb + v) * DDIM + n] = (1.0f - zz[v]) * (u - (float)h8[v]);
    }
  });
}

extern "C" void kernel_launch(void* const* d_in, const int* in_sizes, int n_in,
                              void* d_out, int out_size, void* d_ws, size_t ws_size,
                              hipStream_t stream) {
  const float* h  = (const float*)d_in[0];
  const float* A  = (const float*)d_in[1];
  const float* Wr = (const float*)d_in[2];
  const float* br = (const float*)d_in[3];
  const float* Wz = (const float*)d_in[4];
  const float* bz = (const float*)d_in[5];
  const float* Wh = (const float*)d_in[6];
  const float* bh = (const float*)d_in[7];

  const int B = in_sizes[0] / (NDIM * DDIM);   // 1024
  dim3 grid(B), block(256);
  gcgru_cell_kernel<<<grid, block, 0, stream>>>(h, A, Wr, br, Wz, bz, Wh, bh,
                                                (float*)d_out);
}
